// Attention_10325101380252
// MI455X (gfx1250) — matrix operand order repaired and checked by reading, not compiled, not run on hardware
//
#include <hip/hip_runtime.h>
#include <hip/hip_bf16.h>

typedef __attribute__((ext_vector_type(16))) __bf16 v16bf;
typedef __attribute__((ext_vector_type(8)))  float  v8f;

#define HIDN  1024
#define DH    64
#define NH    16
#define SEQ   2048
#define BATCH 4
#define KCH   64      // keys staged per block round in attention

// ---------------------------------------------------------------------------
// Kernel 0: one-shot fp32 -> bf16 conversion (x and the three W matrices).
// ---------------------------------------------------------------------------
__global__ void cvt_bf16_kernel(const float* __restrict__ in,
                                __bf16* __restrict__ out, int n4)
{
    int i = blockIdx.x * blockDim.x + threadIdx.x;
    if (i >= n4) return;
    float4 v = ((const float4*)in)[i];
    out[4 * i + 0] = (__bf16)v.x;
    out[4 * i + 1] = (__bf16)v.y;
    out[4 * i + 2] = (__bf16)v.z;
    out[4 * i + 3] = (__bf16)v.w;
}

// ---------------------------------------------------------------------------
// Kernel 1: QKV projection GEMM on pre-converted bf16.  out = x @ W^T + b,
// bf16 out in [B,H,S,DH] (q,k) or [B,H,DH,S] (v).  Wave tile: 32(M) x 64(N),
// K stepped by 32 -> 8 wmma per k-step, pure b128 loads (no cvt in loop).
// ---------------------------------------------------------------------------
__global__ __launch_bounds__(128) void qkv_gemm_kernel(
    const __bf16* __restrict__ x,    // [B*S, HIDN] bf16
    const __bf16* __restrict__ W,    // [HIDN, HIDN] bf16, row = out feature
    const float*  __restrict__ bias, // [HIDN] fp32
    __bf16* __restrict__ out,
    int transposed)
{
    const int lane  = threadIdx.x & 31;
    const int wave  = threadIdx.x >> 5;
    const int mbase = blockIdx.x * 32;
    const int nbase = (blockIdx.y * 4 + wave) * 64;
    const int lm    = lane & 15;
    const int kbA   = (lane < 16) ? 0 : 8;    // unused; A-frag uses kbB (contiguous), same as B
    const int kbB   = (lane < 16) ? 0 : 16;   // B-frag K base (contiguous halves)

    v8f acc[2][4] = {};

    for (int ks = 0; ks < HIDN / 32; ++ks) {
        const int koff = ks * 32;
        v16bf a[2];
        #pragma unroll
        for (int mt = 0; mt < 2; ++mt) {
            const __bf16* xrow = x + (size_t)(mbase + mt * 16 + lm) * HIDN + koff;
            #pragma unroll
            for (int j = 0; j < 8; ++j) {
                int k0 = kbB + 2 * j;
                a[mt][2 * j]     = xrow[k0];
                a[mt][2 * j + 1] = xrow[k0 + 1];
            }
        }
        #pragma unroll
        for (int nt = 0; nt < 4; ++nt) {
            const __bf16* wrow = W + (size_t)(nbase + nt * 16 + lm) * HIDN + koff + kbB;
            v16bf bfrag;
            #pragma unroll
            for (int e = 0; e < 16; ++e) bfrag[e] = wrow[e];
            acc[0][nt] = __builtin_amdgcn_wmma_f32_16x16x32_bf16(
                false, a[0], false, bfrag, (short)0, acc[0][nt], false, false);
            acc[1][nt] = __builtin_amdgcn_wmma_f32_16x16x32_bf16(
                false, a[1], false, bfrag, (short)0, acc[1][nt], false, false);
        }
    }

    const int halfsel = (lane < 16) ? 0 : 8;
    #pragma unroll
    for (int mt = 0; mt < 2; ++mt) {
        #pragma unroll
        for (int nt = 0; nt < 4; ++nt) {
            int n = nbase + nt * 16 + lm;
            float bv = bias[n];
            int h = n >> 6, d = n & 63;
            #pragma unroll
            for (int r = 0; r < 8; ++r) {
                int t = mbase + mt * 16 + r + halfsel;
                int bidx = t / SEQ, s = t - bidx * SEQ;
                float val = acc[mt][nt][r] + bv;
                size_t oidx = transposed
                    ? ((size_t)(bidx * NH + h) * DH + d) * SEQ + s
                    : ((size_t)(bidx * NH + h) * SEQ + s) * DH + d;
                out[oidx] = (__bf16)val;
            }
        }
    }
}

// ---------------------------------------------------------------------------
// Kernel 2: RoPE on q and k.  sinusoidal_pos[..., :32]=sin, [..., 32:]=cos.
// ---------------------------------------------------------------------------
__global__ void rope_kernel(__bf16* __restrict__ q, __bf16* __restrict__ k,
                            const float* __restrict__ sp, int total)
{
    int idx = blockIdx.x * blockDim.x + threadIdx.x;
    if (idx >= total) return;                 // total = B*H*S*32
    int d2 = idx & 31;
    int s  = (idx >> 5) & (SEQ - 1);
    size_t base = ((size_t)(idx >> 5)) * DH + 2 * d2;
    float sinv = sp[(size_t)s * DH + d2];
    float cosv = sp[(size_t)s * DH + 32 + d2];
    float q0 = (float)q[base], q1 = (float)q[base + 1];
    q[base]     = (__bf16)(q0 * cosv - q1 * sinv);
    q[base + 1] = (__bf16)(q1 * cosv + q0 * sinv);
    float k0 = (float)k[base], k1 = (float)k[base + 1];
    k[base]     = (__bf16)(k0 * cosv - k1 * sinv);
    k[base + 1] = (__bf16)(k1 * cosv + k0 * sinv);
}

// ---------------------------------------------------------------------------
// Kernel 3: flash attention with LDS-staged K/V tiles.
// Block = 8 waves x 32 queries = 256 queries.  Per round, the block stages a
// 64-key K tile and V tile into LDS (shared by all 8 waves), prefetches the
// next tile, then each wave runs two 32-key sub-chunks of online softmax:
// QK^T 8 wmma + PV 8 wmma per sub-chunk.
// ---------------------------------------------------------------------------
__global__ __launch_bounds__(256) void attn_kernel(
    const __bf16* __restrict__ qb,   // [B,H,S,DH]
    const __bf16* __restrict__ kb,   // [B,H,S,DH]
    const __bf16* __restrict__ vb,   // [B,H,DH,S]
    const float*  __restrict__ amask,// [B,S]
    float* __restrict__ out)         // [B,S,HIDN]
{
    __shared__ __align__(16) __bf16 ldsK[KCH][DH];        // 8 KB
    __shared__ __align__(16) __bf16 ldsV[DH][KCH];        // 8 KB
    __shared__ __align__(16) __bf16 lds_p[8][2][16][32];  // 16 KB

    const int tid   = threadIdx.x;
    const int lane  = tid & 31;
    const int wave  = tid >> 5;
    const int bh    = blockIdx.y;
    const int b     = bh >> 4, h = bh & 15;
    const int qbase = blockIdx.x * 256 + wave * 32;       // 2 q-tiles per wave
    const int lm    = lane & 15;
    const int kbA   = (lane < 16) ? 0 : 8;
    const int kbB   = (lane < 16) ? 0 : 16;
    const int halfsel = (lane < 16) ? 0 : 8;

    const __bf16* qhead = qb + (size_t)bh * SEQ * DH;
    const __bf16* khead = kb + (size_t)bh * SEQ * DH;
    const __bf16* vhead = vb + (size_t)bh * DH * SEQ;
    const float*  mrow  = amask + (size_t)b * SEQ;

    // Q A-fragments, resident for whole kernel: [m-tile][k-step]
    v16bf qa[2][2];
    #pragma unroll
    for (int mt = 0; mt < 2; ++mt) {
        const __bf16* qrow = qhead + (size_t)(qbase + mt * 16 + lm) * DH;
        #pragma unroll
        for (int ks = 0; ks < 2; ++ks)
            #pragma unroll
            for (int j = 0; j < 8; ++j) {
                int k0 = ks * 32 + kbB + 2 * j;
                qa[mt][ks][2 * j]     = qrow[k0];
                qa[mt][ks][2 * j + 1] = qrow[k0 + 1];
            }
    }

    v8f acc[2][4] = {};
    float mstat[2][8], lstat[2][8];
    #pragma unroll
    for (int mt = 0; mt < 2; ++mt)
        #pragma unroll
        for (int r = 0; r < 8; ++r) { mstat[mt][r] = -1e30f; lstat[mt][r] = 0.f; }

    const float scale = 0.125f;                 // 1/sqrt(DH)
    const float LOG2E = 1.4426950408889634f;

    for (int round = 0; round < SEQ / KCH; ++round) {
        const int key0r = round * KCH;

        // Stage K (64x64) and V^T (64x64) tiles: 512 x 16B each, 256 threads.
        #pragma unroll
        for (int e = tid; e < 512; e += 256) {
            int row = e >> 3, seg = (e & 7) * 8;
            *(uint4*)&ldsK[row][seg] =
                *(const uint4*)(khead + (size_t)(key0r + row) * DH + seg);
            *(uint4*)&ldsV[row][seg] =
                *(const uint4*)(vhead + (size_t)row * SEQ + key0r + seg);
        }
        // Prefetch next round's tiles into cache (global_prefetch_b8).
        if (round + 1 < SEQ / KCH) {
            int nk = key0r + KCH;
            #pragma unroll
            for (int e = tid; e < 512; e += 256) {
                int row = e >> 3, seg = (e & 7) * 8;
                __builtin_prefetch(khead + (size_t)(nk + row) * DH + seg, 0, 0);
                __builtin_prefetch(vhead + (size_t)row * SEQ + nk + seg, 0, 0);
            }
        }
        __syncthreads();

        #pragma unroll
        for (int sc = 0; sc < 2; ++sc) {
            const int key0 = key0r + sc * 32;

            // QK^T: 2 n-tiles x 2 k-steps, K frags shared by both m-tiles.
            v8f s[2][2] = {};
            #pragma unroll
            for (int nt = 0; nt < 2; ++nt)
                #pragma unroll
                for (int ks = 0; ks < 2; ++ks) {
                    const __bf16* kr = &ldsK[sc * 32 + nt * 16 + lm][ks * 32 + kbB];
                    v16bf kf;
                    #pragma unroll
                    for (int e = 0; e < 16; ++e) kf[e] = kr[e];
                    s[0][nt] = __builtin_amdgcn_wmma_f32_16x16x32_bf16(
                        false, qa[0][ks], false, kf, (short)0, s[0][nt], false, false);
                    s[1][nt] = __builtin_amdgcn_wmma_f32_16x16x32_bf16(
                        false, qa[1][ks], false, kf, (short)0, s[1][nt], false, false);
                }

            const float mv0 = mrow[key0 + lm];
            const float mv1 = mrow[key0 + 16 + lm];

            // Online softmax; a row lives in 16 lanes of one half-wave.
            #pragma unroll
            for (int mt = 0; mt < 2; ++mt)
                #pragma unroll
                for (int r = 0; r < 8; ++r) {
                    float a0 = s[mt][0][r] * scale + mv0;
                    float a1 = s[mt][1][r] * scale + mv1;
                    float vmax = fmaxf(a0, a1);
                    vmax = fmaxf(vmax, __shfl_xor(vmax, 1, 32));
                    vmax = fmaxf(vmax, __shfl_xor(vmax, 2, 32));
                    vmax = fmaxf(vmax, __shfl_xor(vmax, 4, 32));
                    vmax = fmaxf(vmax, __shfl_xor(vmax, 8, 32));
                    float newm  = fmaxf(mstat[mt][r], vmax);
                    float alpha = exp2f((mstat[mt][r] - newm) * LOG2E);
                    float p0 = exp2f((a0 - newm) * LOG2E);
                    float p1 = exp2f((a1 - newm) * LOG2E);
                    float vsum = p0 + p1;
                    vsum += __shfl_xor(vsum, 1, 32);
                    vsum += __shfl_xor(vsum, 2, 32);
                    vsum += __shfl_xor(vsum, 4, 32);
                    vsum += __shfl_xor(vsum, 8, 32);
                    lstat[mt][r] = lstat[mt][r] * alpha + vsum;
                    mstat[mt][r] = newm;
                    #pragma unroll
                    for (int dt = 0; dt < 4; ++dt) acc[mt][dt][r] *= alpha;
                    int m = r + halfsel;
                    lds_p[wave][mt][m][lm]      = (__bf16)p0;
                    lds_p[wave][mt][m][16 + lm] = (__bf16)p1;
                }

            // In-wave DS ordering for the D-layout -> A-layout P transpose.
            asm volatile("s_wait_dscnt 0" ::: "memory");

            // PV: per m-tile, P A-frag (16x32) x 4 V B-frags from LDS.
            #pragma unroll
            for (int mt = 0; mt < 2; ++mt) {
                v16bf pa;
                #pragma unroll
                for (int j = 0; j < 8; ++j) {
                    int k0 = kbB + 2 * j;
                    pa[2 * j]     = lds_p[wave][mt][lm][k0];
                    pa[2 * j + 1] = lds_p[wave][mt][lm][k0 + 1];
                }
                #pragma unroll
                for (int dt = 0; dt < 4; ++dt) {
                    const __bf16* vr = &ldsV[dt * 16 + lm][sc * 32 + kbB];
                    v16bf vf;
                    #pragma unroll
                    for (int e = 0; e < 16; ++e) vf[e] = vr[e];
                    acc[mt][dt] = __builtin_amdgcn_wmma_f32_16x16x32_bf16(
                        false, pa, false, vf, (short)0, acc[mt][dt], false, false);
                }
            }
        }
        __syncthreads();
    }

    // Normalize and write ctx [B,S,H*DH] fp32.
    #pragma unroll
    for (int mt = 0; mt < 2; ++mt)
        #pragma unroll
        for (int r = 0; r < 8; ++r) {
            float inv = 1.0f / lstat[mt][r];
            int sq = qbase + mt * 16 + r + halfsel;
            float* orow = out + ((size_t)b * SEQ + sq) * HIDN + h * DH;
            #pragma unroll
            for (int dt = 0; dt < 4; ++dt)
                orow[dt * 16 + lm] = acc[mt][dt][r] * inv;
        }
}

// ---------------------------------------------------------------------------
extern "C" void kernel_launch(void* const* d_in, const int* in_sizes, int n_in,
                              void* d_out, int out_size, void* d_ws, size_t ws_size,
                              hipStream_t stream)
{
    const float* x  = (const float*)d_in[0];
    const float* am = (const float*)d_in[1];
    const float* sp = (const float*)d_in[2];
    const float* Wq = (const float*)d_in[3];
    const float* bq = (const float*)d_in[4];
    const float* Wk = (const float*)d_in[5];
    const float* bk = (const float*)d_in[6];
    const float* Wv = (const float*)d_in[7];
    const float* bv = (const float*)d_in[8];
    float* out = (float*)d_out;

    char* ws = (char*)d_ws;
    __bf16* qbuf = (__bf16*)(ws);                            // 16 MB
    __bf16* kbuf = (__bf16*)(ws + ((size_t)16 << 20));       // 16 MB
    __bf16* vbuf = (__bf16*)(ws + ((size_t)32 << 20));       // 16 MB
    __bf16* xb   = (__bf16*)(ws + ((size_t)48 << 20));       // 16 MB
    __bf16* wqb  = (__bf16*)(ws + ((size_t)64 << 20));       // 2 MB
    __bf16* wkb  = (__bf16*)(ws + ((size_t)66 << 20));       // 2 MB
    __bf16* wvb  = (__bf16*)(ws + ((size_t)68 << 20));       // 2 MB

    int xe4 = BATCH * SEQ * HIDN / 4;
    int we4 = HIDN * HIDN / 4;
    cvt_bf16_kernel<<<(xe4 + 255) / 256, 256, 0, stream>>>(x,  xb,  xe4);
    cvt_bf16_kernel<<<(we4 + 255) / 256, 256, 0, stream>>>(Wq, wqb, we4);
    cvt_bf16_kernel<<<(we4 + 255) / 256, 256, 0, stream>>>(Wk, wkb, we4);
    cvt_bf16_kernel<<<(we4 + 255) / 256, 256, 0, stream>>>(Wv, wvb, we4);

    dim3 gg((BATCH * SEQ) / 32, HIDN / 256, 1);
    qkv_gemm_kernel<<<gg, 128, 0, stream>>>(xb, wqb, bq, qbuf, 0);
    qkv_gemm_kernel<<<gg, 128, 0, stream>>>(xb, wkb, bk, kbuf, 0);
    qkv_gemm_kernel<<<gg, 128, 0, stream>>>(xb, wvb, bv, vbuf, 1);

    int pairs = BATCH * NH * SEQ * (DH / 2);
    rope_kernel<<<(pairs + 255) / 256, 256, 0, stream>>>(qbuf, kbuf, sp, pairs);

    dim3 ga(SEQ / 256, BATCH * NH, 1);   // 8 waves/block, 32 queries/wave
    attn_kernel<<<ga, 256, 0, stream>>>(qbuf, kbuf, vbuf, am, out);
}
